// GAT_11605001633834
// MI455X (gfx1250) — matrix operand order repaired and checked by reading, not compiled, not run on hardware
//
#include <hip/hip_runtime.h>
#include <hip/hip_bf16.h>
#include <stdint.h>

// Problem constants (from reference)
#define NN    20000
#define EE    320000
#define HH    8
#define CC    32
#define HC    256      // H*C
#define GG    64
#define FIN   41
#define KP1   64       // F_IN padded to multiple of 32 for WMMA K
#define NEDGE (EE + NN)  // edges + self loops
#define NEG_SLOPE 0.2f
#define BN_EPS    1e-5f

typedef __attribute__((ext_vector_type(16))) __bf16   v16bf;
typedef __attribute__((ext_vector_type(8)))  float    v8f;
typedef __attribute__((ext_vector_type(4)))  uint32_t u32x4;
typedef __attribute__((ext_vector_type(4)))  float    f32x4;

// ---------- helpers ----------
__device__ __forceinline__ uint16_t f2bf(float f) {
    uint32_t u = __float_as_uint(f);
    u += 0x7FFFu + ((u >> 16) & 1u);   // round-to-nearest-even
    return (uint16_t)(u >> 16);
}

__device__ __forceinline__ void atomicMaxF(float* addr, float val) {
    // classic bit-trick: works with -inf initialization
    if (val >= 0.0f) atomicMax((int*)addr, __float_as_int(val));
    else             atomicMin((unsigned int*)addr, __float_as_uint(val));
}

__device__ __forceinline__ float lrelu(float v) {
    return v > 0.0f ? v : NEG_SLOPE * v;
}

// ---------- generic fill ----------
__global__ void fill_f32(float* p, long n, float v) {
    long i = (long)blockIdx.x * blockDim.x + threadIdx.x;
    if (i < n) p[i] = v;
}

// ---------- conversions ----------
// src [rows][cols] f32 row-major -> dst [rows][colsp] bf16, zero padded
__global__ void cvt_pad_rows(const float* __restrict__ src, unsigned short* __restrict__ dst,
                             int rows, int cols, int colsp) {
    long i = (long)blockIdx.x * blockDim.x + threadIdx.x;
    if (i >= (long)rows * colsp) return;
    int r = (int)(i / colsp), c = (int)(i % colsp);
    float v = (c < cols) ? src[(long)r * cols + c] : 0.0f;
    dst[i] = f2bf(v);
}

// W [K][Ntot] f32 row-major -> Wt [Ntot][Kp] bf16 (transposed, zero padded)
__global__ void cvt_transpose(const float* __restrict__ W, unsigned short* __restrict__ Wt,
                              int K, int Ntot, int Kp) {
    long i = (long)blockIdx.x * blockDim.x + threadIdx.x;
    if (i >= (long)Ntot * Kp) return;
    int n = (int)(i / Kp), k = (int)(i % Kp);
    float v = (k < K) ? W[(long)k * Ntot + n] : 0.0f;
    Wt[i] = f2bf(v);
}

// ---------- WMMA bf16 GEMM: C[M][Ntot] = A[M][Kp] * Bt[Ntot][Kp]^T ----------
// One wave per 16x16 output tile. M%16==0, Ntot%16==0, Kp%32==0 required.
__global__ void gemm_bf16_wmma(const unsigned short* __restrict__ A,
                               const unsigned short* __restrict__ Bt,
                               float* __restrict__ C,
                               int M, int Kp, int Ntot) {
    int wave  = (int)((blockIdx.x * blockDim.x + threadIdx.x) >> 5);
    int lane  = threadIdx.x & 31;
    int ntile = Ntot >> 4;
    int tm = wave / ntile, tn = wave % ntile;
    int row0 = tm << 4, col0 = tn << 4;
    int lhalf = lane >> 4;   // 0: lanes 0-15, 1: lanes 16-31
    int l     = lane & 15;

    const unsigned short* arow = A  + (size_t)(row0 + l) * Kp;  // row-major A row
    const unsigned short* bcol = Bt + (size_t)(col0 + l) * Kp;  // column of B (row of Bt)

    v8f acc = {};
    for (int k0 = 0; k0 < Kp; k0 += 32) {
        union { v16bf v; u32x4 u[2]; } a, b;
        // A fragment (16-bit, 16x32): lanes 0-15 hold K = k0+[0..7] and k0+[16..23];
        // lanes 16-31 hold K = k0+[8..15] and k0+[24..31].
        int ka = k0 + lhalf * 8;
        a.u[0] = *(const u32x4*)(arow + ka);
        a.u[1] = *(const u32x4*)(arow + ka + 16);
        // B fragment (32x16): lanes 0-15 col=col0+l K = k0+[0..7] and k0+[16..23];
        // lanes 16-31 same col K = k0+[8..15] and k0+[24..31].
        int kb = k0 + lhalf * 8;
        b.u[0] = *(const u32x4*)(bcol + kb);
        b.u[1] = *(const u32x4*)(bcol + kb + 16);

        if (k0 + 32 < Kp) {  // gfx1250 global_prefetch_b8 for next K-chunk
            __builtin_prefetch(arow + ka + 32, 0, 0);
            __builtin_prefetch(bcol + kb + 32, 0, 0);
        }
        acc = __builtin_amdgcn_wmma_f32_16x16x32_bf16(
            /*neg_a=*/false, a.v, /*neg_b=*/false, b.v,
            /*c_mod=*/(short)0, acc, /*reuse_a=*/false, /*reuse_b=*/false);
    }
    // f32 C/D layout: VGPR i -> (M = i + 8*lhalf, N = l)
    int rbase = row0 + lhalf * 8;
#pragma unroll
    for (int i = 0; i < 8; ++i)
        C[(size_t)(rbase + i) * Ntot + col0 + l] = acc[i];
}

// ---------- attention scores: s_src/s_dst [N][H] ----------
__global__ void scores_kernel(const float* __restrict__ h,
                              const float* __restrict__ a_src, const float* __restrict__ a_dst,
                              float* __restrict__ s_src, float* __restrict__ s_dst) {
    int idx = blockIdx.x * blockDim.x + threadIdx.x;
    if (idx >= NN * HH) return;
    int n = idx / HH, hh = idx % HH;
    const float* hp = h + (size_t)n * HC + hh * CC;
    float ss = 0.0f, sd = 0.0f;
#pragma unroll 8
    for (int c = 0; c < CC; ++c) {
        float v = hp[c];
        ss += v * a_src[hh * CC + c];
        sd += v * a_dst[hh * CC + c];
    }
    s_src[idx] = ss;
    s_dst[idx] = sd;
}

__device__ __forceinline__ void edge_ends(const int* ei, int e, int& s, int& d) {
    if (e < EE) { s = ei[e]; d = ei[EE + e]; }
    else        { s = d = e - EE; }           // self loop
}

// ---------- pass 1: segment max over destinations ----------
__global__ void edge_max(const int* __restrict__ ei,
                         const float* __restrict__ s_src, const float* __restrict__ s_dst,
                         float* __restrict__ segmax) {
    long idx = (long)blockIdx.x * blockDim.x + threadIdx.x;
    if (idx >= (long)NEDGE * HH) return;
    int e = (int)(idx >> 3), hh = (int)(idx & 7);
    int s, d; edge_ends(ei, e, s, d);
    float sc = lrelu(s_src[s * HH + hh] + s_dst[d * HH + hh]);
    atomicMaxF(&segmax[d * HH + hh], sc);
}

// ---------- pass 2: segment sum of exp(score - max) ----------
__global__ void edge_sum(const int* __restrict__ ei,
                         const float* __restrict__ s_src, const float* __restrict__ s_dst,
                         const float* __restrict__ segmax, float* __restrict__ segsum) {
    long idx = (long)blockIdx.x * blockDim.x + threadIdx.x;
    if (idx >= (long)NEDGE * HH) return;
    int e = (int)(idx >> 3), hh = (int)(idx & 7);
    int s, d; edge_ends(ei, e, s, d);
    float sc = lrelu(s_src[s * HH + hh] + s_dst[d * HH + hh]);
    atomicAdd(&segsum[d * HH + hh], expf(sc - segmax[d * HH + hh]));
}

// ---------- pass 3: alpha-weighted scatter of source features ----------
__global__ void edge_scatter(const int* __restrict__ ei,
                             const float* __restrict__ s_src, const float* __restrict__ s_dst,
                             const float* __restrict__ segmax, const float* __restrict__ segsum,
                             const float* __restrict__ h, float* __restrict__ out) {
    long idx = (long)blockIdx.x * blockDim.x + threadIdx.x;
    if (idx >= (long)NEDGE * HH) return;
    int e = (int)(idx >> 3), hh = (int)(idx & 7);
    int s, d; edge_ends(ei, e, s, d);
    float sc    = lrelu(s_src[s * HH + hh] + s_dst[d * HH + hh]);
    float alpha = expf(sc - segmax[d * HH + hh]) / segsum[d * HH + hh];
    const f32x4* hs = (const f32x4*)(h + (size_t)s * HC + hh * CC);
    float* o = out + (size_t)d * HC + hh * CC;
#pragma unroll
    for (int c4 = 0; c4 < CC / 4; ++c4) {
        f32x4 v = hs[c4];
        atomicAdd(&o[c4 * 4 + 0], alpha * v.x);
        atomicAdd(&o[c4 * 4 + 1], alpha * v.y);
        atomicAdd(&o[c4 * 4 + 2], alpha * v.z);
        atomicAdd(&o[c4 * 4 + 3], alpha * v.w);
    }
}

// ---------- bias + eval-BN + ELU (+ optional bf16 copy for next GEMM) ----------
__global__ void finalize_layer(const float* __restrict__ agg, const float* __restrict__ bias,
                               const float* __restrict__ gamma, const float* __restrict__ beta,
                               float* __restrict__ act, unsigned short* __restrict__ actb) {
    long i = (long)blockIdx.x * blockDim.x + threadIdx.x;
    if (i >= (long)NN * HC) return;
    int j = (int)(i & (HC - 1));
    const float inv = rsqrtf(1.0f + BN_EPS);
    float v = agg[i] + bias[j];
    v = v * (gamma[j] * inv) + beta[j];
    v = v > 0.0f ? v : (expf(v) - 1.0f);   // ELU
    act[i] = v;
    if (actb) actb[i] = f2bf(v);
}

// ---------- global mean/max pooling per graph ----------
__global__ void pool_kernel(const float* __restrict__ h, const int* __restrict__ batch,
                            float* __restrict__ mean_acc, float* __restrict__ max_acc,
                            float* __restrict__ cnt) {
    long i = (long)blockIdx.x * blockDim.x + threadIdx.x;
    if (i >= (long)NN * HC) return;
    int n = (int)(i >> 8), j = (int)(i & (HC - 1));
    int g = batch[n];
    float v = h[i];
    atomicAdd(&mean_acc[g * HC + j], v);
    atomicMaxF(&max_acc[g * HC + j], v);
    if (j == 0) atomicAdd(&cnt[g], 1.0f);
}

// ---------- MLP head ----------
__global__ void mlp1_kernel(const float* __restrict__ mean_acc, const float* __restrict__ max_acc,
                            const float* __restrict__ cnt, const float* __restrict__ gsz,
                            const float* __restrict__ w, const float* __restrict__ b,
                            const float* __restrict__ gamma, const float* __restrict__ beta,
                            float* __restrict__ z1) {
    int g = blockIdx.x, j = threadIdx.x;   // 64 graphs x 64 units
    float invc = 1.0f / fmaxf(cnt[g], 1.0f);
    float acc = b[j];
    for (int i = 0; i < HC; ++i) acc += mean_acc[g * HC + i] * invc * w[i * 64 + j];
    for (int i = 0; i < HC; ++i) acc += max_acc[g * HC + i] * w[(HC + i) * 64 + j];
    acc += gsz[g * 2 + 0] * w[(2 * HC + 0) * 64 + j];
    acc += gsz[g * 2 + 1] * w[(2 * HC + 1) * 64 + j];
    const float inv = rsqrtf(1.0f + BN_EPS);
    acc = acc * (gamma[j] * inv) + beta[j];
    z1[g * 64 + j] = fmaxf(acc, 0.0f);
}

__global__ void mlp2_kernel(const float* __restrict__ z1, const float* __restrict__ w,
                            const float* __restrict__ b, const float* __restrict__ gamma,
                            const float* __restrict__ beta, float* __restrict__ z2) {
    int g = blockIdx.x, j = threadIdx.x;   // 64 graphs x 32 units
    float acc = b[j];
    for (int i = 0; i < 64; ++i) acc += z1[g * 64 + i] * w[i * 32 + j];
    const float inv = rsqrtf(1.0f + BN_EPS);
    acc = acc * (gamma[j] * inv) + beta[j];
    z2[g * 32 + j] = fmaxf(acc, 0.0f);
}

__global__ void mlp3_kernel(const float* __restrict__ z2, const float* __restrict__ w,
                            const float* __restrict__ b, float* __restrict__ out) {
    int g = blockIdx.x, j = threadIdx.x;   // 64 graphs x 2 logits
    float acc = b[j];
    for (int i = 0; i < 32; ++i) acc += z2[g * 32 + i] * w[i * 2 + j];
    out[g * 2 + j] = acc;
}

// =====================================================================
extern "C" void kernel_launch(void* const* d_in, const int* in_sizes, int n_in,
                              void* d_out, int out_size, void* d_ws, size_t ws_size,
                              hipStream_t stream) {
    const float* x      = (const float*)d_in[0];
    const int*   ei     = (const int*)  d_in[1];
    const int*   batch  = (const int*)  d_in[2];
    const float* gsz    = (const float*)d_in[3];
    const float* W1     = (const float*)d_in[4];
    const float* a_src1 = (const float*)d_in[5];
    const float* a_dst1 = (const float*)d_in[6];
    const float* b1     = (const float*)d_in[7];
    const float* g1     = (const float*)d_in[8];
    const float* be1    = (const float*)d_in[9];
    const float* W2     = (const float*)d_in[10];
    const float* a_src2 = (const float*)d_in[11];
    const float* a_dst2 = (const float*)d_in[12];
    const float* b2     = (const float*)d_in[13];
    const float* g2     = (const float*)d_in[14];
    const float* be2    = (const float*)d_in[15];
    const float* fc1_w  = (const float*)d_in[16];
    const float* fc1_b  = (const float*)d_in[17];
    const float* gf1    = (const float*)d_in[18];
    const float* bf1    = (const float*)d_in[19];
    const float* fc2_w  = (const float*)d_in[20];
    const float* fc2_b  = (const float*)d_in[21];
    const float* gf2    = (const float*)d_in[22];
    const float* bf2    = (const float*)d_in[23];
    const float* fc3_w  = (const float*)d_in[24];
    const float* fc3_b  = (const float*)d_in[25];
    float* out = (float*)d_out;

    // ---- workspace carve-up (256B aligned) ----
    size_t off = 0;
    auto carve = [&](size_t bytes) -> void* {
        void* p = (char*)d_ws + off;
        off += (bytes + 255) & ~(size_t)255;
        return p;
    };
    unsigned short* xb    = (unsigned short*)carve((size_t)NN * KP1 * 2);
    unsigned short* w1t   = (unsigned short*)carve((size_t)HC * KP1 * 2);
    unsigned short* w2t   = (unsigned short*)carve((size_t)HC * HC  * 2);
    float* featpre        = (float*)carve((size_t)NN * HC * 4);
    float* agg            = (float*)carve((size_t)NN * HC * 4);
    float* act            = (float*)carve((size_t)NN * HC * 4);
    unsigned short* actb  = (unsigned short*)carve((size_t)NN * HC * 2);
    float* s_src          = (float*)carve((size_t)NN * HH * 4);
    float* s_dst          = (float*)carve((size_t)NN * HH * 4);
    float* segmax         = (float*)carve((size_t)NN * HH * 4);
    float* segsum         = (float*)carve((size_t)NN * HH * 4);
    float* mean_acc       = (float*)carve((size_t)GG * HC * 4);
    float* max_acc        = (float*)carve((size_t)GG * HC * 4);
    float* cnt            = (float*)carve((size_t)GG * 4);
    float* z1             = (float*)carve((size_t)GG * 64 * 4);
    float* z2             = (float*)carve((size_t)GG * 32 * 4);

    const int B = 256;
    auto blocks = [](long n, int b) { return (int)((n + b - 1) / b); };
    const float NEG_INF = -__builtin_huge_valf();

    // ---- conversions ----
    cvt_pad_rows <<<blocks((long)NN * KP1, B), B, 0, stream>>>(x,  xb,  NN, FIN, KP1);
    cvt_transpose<<<blocks((long)HC * KP1, B), B, 0, stream>>>(W1, w1t, FIN, HC, KP1);
    cvt_transpose<<<blocks((long)HC * HC,  B), B, 0, stream>>>(W2, w2t, HC,  HC, HC);

    const long edgeThreads = (long)NEDGE * HH;
    const long featN       = (long)NN * HC;

    // ============== GAT layer 1 ==============
    gemm_bf16_wmma<<<(NN / 16) * (HC / 16) / 8, B, 0, stream>>>(xb, w1t, featpre, NN, KP1, HC);
    scores_kernel <<<blocks((long)NN * HH, B), B, 0, stream>>>(featpre, a_src1, a_dst1, s_src, s_dst);
    fill_f32<<<blocks((long)NN * HH, B), B, 0, stream>>>(segmax, (long)NN * HH, NEG_INF);
    fill_f32<<<blocks((long)NN * HH, B), B, 0, stream>>>(segsum, (long)NN * HH, 0.0f);
    fill_f32<<<blocks(featN, B), B, 0, stream>>>(agg, featN, 0.0f);
    edge_max    <<<blocks(edgeThreads, B), B, 0, stream>>>(ei, s_src, s_dst, segmax);
    edge_sum    <<<blocks(edgeThreads, B), B, 0, stream>>>(ei, s_src, s_dst, segmax, segsum);
    edge_scatter<<<blocks(edgeThreads, B), B, 0, stream>>>(ei, s_src, s_dst, segmax, segsum, featpre, agg);
    finalize_layer<<<blocks(featN, B), B, 0, stream>>>(agg, b1, g1, be1, act, actb);

    // ============== GAT layer 2 ==============
    gemm_bf16_wmma<<<(NN / 16) * (HC / 16) / 8, B, 0, stream>>>(actb, w2t, featpre, NN, HC, HC);
    scores_kernel <<<blocks((long)NN * HH, B), B, 0, stream>>>(featpre, a_src2, a_dst2, s_src, s_dst);
    fill_f32<<<blocks((long)NN * HH, B), B, 0, stream>>>(segmax, (long)NN * HH, NEG_INF);
    fill_f32<<<blocks((long)NN * HH, B), B, 0, stream>>>(segsum, (long)NN * HH, 0.0f);
    fill_f32<<<blocks(featN, B), B, 0, stream>>>(agg, featN, 0.0f);
    edge_max    <<<blocks(edgeThreads, B), B, 0, stream>>>(ei, s_src, s_dst, segmax);
    edge_sum    <<<blocks(edgeThreads, B), B, 0, stream>>>(ei, s_src, s_dst, segmax, segsum);
    edge_scatter<<<blocks(edgeThreads, B), B, 0, stream>>>(ei, s_src, s_dst, segmax, segsum, featpre, agg);
    finalize_layer<<<blocks(featN, B), B, 0, stream>>>(agg, b2, g2, be2, act, (unsigned short*)nullptr);

    // ============== pooling + MLP ==============
    fill_f32<<<blocks((long)GG * HC, B), B, 0, stream>>>(mean_acc, (long)GG * HC, 0.0f);
    fill_f32<<<blocks((long)GG * HC, B), B, 0, stream>>>(max_acc,  (long)GG * HC, NEG_INF);
    fill_f32<<<1, GG, 0, stream>>>(cnt, (long)GG, 0.0f);
    pool_kernel<<<blocks(featN, B), B, 0, stream>>>(act, batch, mean_acc, max_acc, cnt);

    mlp1_kernel<<<GG, 64, 0, stream>>>(mean_acc, max_acc, cnt, gsz, fc1_w, fc1_b, gf1, bf1, z1);
    mlp2_kernel<<<GG, 32, 0, stream>>>(z1, fc2_w, fc2_b, gf2, bf2, z2);
    mlp3_kernel<<<GG, 2, 0, stream>>>(z2, fc3_w, fc3_b, out);

    (void)in_sizes; (void)n_in; (void)out_size; (void)ws_size;
}
